// Seq2seq_3616362463398
// MI455X (gfx1250) — matrix operand order repaired and checked by reading, not compiled, not run on hardware
//
#include <hip/hip_runtime.h>
#include <hip/hip_bf16.h>
#include <math.h>

typedef __bf16 bhalf;
typedef __attribute__((ext_vector_type(16))) __bf16 v16bf;
typedef __attribute__((ext_vector_type(8)))  __bf16 v8bf;
typedef __attribute__((ext_vector_type(8)))  float  v8f;

constexpr int kB    = 32;     // batch
constexpr int kSenc = 64;
constexpr int kSdec = 32;
constexpr int kE    = 512;
constexpr int kH    = 512;
constexpr int kV    = 32000;
constexpr int kG4H  = 2048;   // 4*H gate width

union V16u { v16bf v; v8bf h[2]; };

// ---- WMMA tile loaders -------------------------------------------------
// A (activations) tile 16x32 bf16, row-major source [M, lda]:
//   lanes 0-15 : M=lane,    VGPR0-3 = K[k0..k0+7],   VGPR4-7 = K[k0+16..k0+23]
//   lanes16-31 : M=lane-16, VGPR0-3 = K[k0+8..k0+15],VGPR4-7 = K[k0+24..k0+31]
__device__ __forceinline__ v16bf load_a_tile(const bhalf* A, int lda,
                                             int m0, int k0, int lane) {
  int row = m0 + (lane & 15);
  int kb  = k0 + ((lane >> 4) << 3);
  const bhalf* p = A + (size_t)row * lda + kb;
  V16u r;
  r.h[0] = *(const v8bf*)(p);        // 16B global_load_b128
  r.h[1] = *(const v8bf*)(p + 16);
  return r.v;
}

// B tile 32x16 with B[k,n] = W[n,k], W row-major [N, ldw=K]:
//   lanes 0-15 : N=n0+lane,    K=k0..k0+7, k0+16..k0+23
//   lanes16-31 : N=n0+lane-16, K=k0+8..k0+15, k0+24..k0+31
__device__ __forceinline__ v16bf load_b_tile(const bhalf* W, int ldw,
                                             int n0, int k0, int lane) {
  int n  = n0 + (lane & 15);
  int kb = k0 + ((lane >> 4) << 3);
  const bhalf* p = W + (size_t)n * ldw + kb;
  V16u r;
  r.h[0] = *(const v8bf*)(p);
  r.h[1] = *(const v8bf*)(p + 16);
  return r.v;
}

__device__ __forceinline__ v8f wmma_bf16(v8f c, v16bf a, v16bf b) {
  return __builtin_amdgcn_wmma_f32_16x16x32_bf16(false, a, false, b,
                                                 (short)0, c, false, false);
}

__device__ __forceinline__ float sigmf(float x) { return 1.f / (1.f + __expf(-x)); }

// ---- fp32 -> bf16 conversion (weights resident in L2 as bf16) ----------
__global__ void cvt_bf16(const float* __restrict__ s, bhalf* __restrict__ d, int n) {
  int i = blockIdx.x * blockDim.x + threadIdx.x;
  if (i < n) d[i] = (bhalf)s[i];
}

__global__ void make_neg(const int* __restrict__ xs, float* __restrict__ neg) {
  int i = blockIdx.x * blockDim.x + threadIdx.x;
  if (i < kB * kSenc) neg[i] = (xs[i] > 0) ? 0.f : 1e20f;
}

__global__ void zero_state(float* __restrict__ C, bhalf* __restrict__ Hs) {
  int i = blockIdx.x * blockDim.x + threadIdx.x;
  if (i < kB * kH) { C[i] = 0.f; Hs[i] = (bhalf)0.f; }
}

__global__ void embed_gather(const int* __restrict__ ids,
                             const bhalf* __restrict__ emb,
                             bhalf* __restrict__ out) {
  int row = blockIdx.x;
  int id  = ids[row];
  const bhalf* src = emb + (size_t)id * kE;
  bhalf* dst = out + (size_t)row * kE;
  for (int k = threadIdx.x; k < kE; k += blockDim.x) dst[k] = src[k];
}

// ---- LSTM gate GEMM: G = X@Wih^T + H@Whh^T + bih + bhh -----------------
// One wave per 16-wide N tile; both M tiles (M=32) held in registers so
// every B tile is loaded once and fed to two back-to-back WMMAs.
// N=2048 -> 128 waves = 16 blocks of 256 threads.
__global__ void lstm_gates(const bhalf* __restrict__ X, int ldx,
                           const bhalf* __restrict__ Hst,            // [B,H]
                           const bhalf* __restrict__ Wih, int Kx,    // [4H,Kx]
                           const bhalf* __restrict__ Whh,            // [4H,H]
                           const float* __restrict__ bih,
                           const float* __restrict__ bhh,
                           float* __restrict__ G) {                  // [B,4H]
  int wave = (blockIdx.x * blockDim.x + threadIdx.x) >> 5;
  int lane = threadIdx.x & 31;
  int n0 = (wave & 127) << 4;
  v8f acc0 = {}, acc1 = {};
  for (int k = 0; k < Kx; k += 32) {
    v16bf b = load_b_tile(Wih, Kx, n0, k, lane);
    acc0 = wmma_bf16(acc0, load_a_tile(X, ldx, 0,  k, lane), b);
    acc1 = wmma_bf16(acc1, load_a_tile(X, ldx, 16, k, lane), b);
  }
  for (int k = 0; k < kH; k += 32) {
    v16bf b = load_b_tile(Whh, kH, n0, k, lane);
    acc0 = wmma_bf16(acc0, load_a_tile(Hst, kH, 0,  k, lane), b);
    acc1 = wmma_bf16(acc1, load_a_tile(Hst, kH, 16, k, lane), b);
  }
  int n    = n0 + (lane & 15);
  int mb   = (lane >> 4) << 3;
  float bs = bih[n] + bhh[n];
#pragma unroll
  for (int r = 0; r < 8; ++r) {
    G[(size_t)(mb + r) * kG4H + n]      = acc0[r] + bs;
    G[(size_t)(mb + r + 16) * kG4H + n] = acc1[r] + bs;
  }
}

// ---- generic A@W^T wave-GEMM (no bias); EPI: 0=f32, 1=bf16, 2=tanh->bf16
// One wave per N tile, both M tiles in registers (B loaded once).
template <int EPI>
__global__ void gemm_bt(const bhalf* __restrict__ A, int lda,
                        const bhalf* __restrict__ W, int K,          // ldw=K
                        int Ntiles, void* __restrict__ out, int ldo) {
  int wave = (blockIdx.x * blockDim.x + threadIdx.x) >> 5;
  int lane = threadIdx.x & 31;
  if (wave >= Ntiles) return;              // wave-uniform: EXEC stays all-1s
  int n0 = wave << 4;
  v8f acc0 = {}, acc1 = {};
  for (int k = 0; k < K; k += 32) {
    v16bf b = load_b_tile(W, K, n0, k, lane);
    acc0 = wmma_bf16(acc0, load_a_tile(A, lda, 0,  k, lane), b);
    acc1 = wmma_bf16(acc1, load_a_tile(A, lda, 16, k, lane), b);
  }
  int n  = n0 + (lane & 15);
  int mb = (lane >> 4) << 3;
#pragma unroll
  for (int r = 0; r < 8; ++r) {
    float v0 = acc0[r], v1 = acc1[r];
    if (EPI == 0) {
      ((float*)out)[(size_t)(mb + r) * ldo + n]      = v0;
      ((float*)out)[(size_t)(mb + r + 16) * ldo + n] = v1;
    } else if (EPI == 1) {
      ((bhalf*)out)[(size_t)(mb + r) * ldo + n]      = (bhalf)v0;
      ((bhalf*)out)[(size_t)(mb + r + 16) * ldo + n] = (bhalf)v1;
    } else {
      ((bhalf*)out)[(size_t)(mb + r) * ldo + n]      = (bhalf)tanhf(v0);
      ((bhalf*)out)[(size_t)(mb + r + 16) * ldo + n] = (bhalf)tanhf(v1);
    }
  }
}

// ---- LSTM pointwise: gate combine, state update ------------------------
__global__ void lstm_pointwise(const float* __restrict__ Gt,
                               float* __restrict__ C, bhalf* __restrict__ Hs,
                               bhalf* __restrict__ seq_out, int ld_seq) {
  int i = blockIdx.x * blockDim.x + threadIdx.x;
  if (i >= kB * kH) return;
  int b = i >> 9, n = i & (kH - 1);
  const float* g = Gt + (size_t)b * kG4H;
  float ig = sigmf(g[n]);
  float fg = sigmf(g[kH + n]);
  float gg = tanhf(g[2 * kH + n]);
  float og = sigmf(g[3 * kH + n]);
  float c = fg * C[i] + ig * gg;
  C[i] = c;
  float h = og * tanhf(c);
  Hs[(size_t)b * kH + n] = (bhalf)h;
  if (seq_out) seq_out[(size_t)b * ld_seq + n] = (bhalf)h;
}

// ---- attention: pw = hid . enc_out - neg; softmax; ctx; cat=[xe,ctx] ---
__global__ void attn_ctx(const bhalf* __restrict__ hid,      // [B,H]
                         const bhalf* __restrict__ enc_out,  // [B,Senc,H]
                         const float* __restrict__ neg,      // [B,Senc]
                         const bhalf* __restrict__ yemb_t, int ld_ye,
                         bhalf* __restrict__ cat) {          // [B, E+H]
  int b   = blockIdx.x;
  int tid = threadIdx.x;                   // 64 threads = one per enc step
  __shared__ float aw[kSenc];
  __shared__ float red[kSenc];
  const bhalf* hq = hid + (size_t)b * kH;
  const bhalf* eb = enc_out + (size_t)b * kSenc * kH;
  const bhalf* er = eb + (size_t)tid * kH;
  float acc = 0.f;
  for (int k = 0; k < kH; k += 4)
    acc += (float)hq[k]     * (float)er[k]
         + (float)hq[k + 1] * (float)er[k + 1]
         + (float)hq[k + 2] * (float)er[k + 2]
         + (float)hq[k + 3] * (float)er[k + 3];
  aw[tid]  = acc - neg[b * kSenc + tid];
  red[tid] = aw[tid];
  __syncthreads();
  for (int s = 32; s > 0; s >>= 1) {
    if (tid < s) red[tid] = fmaxf(red[tid], red[tid + s]);
    __syncthreads();
  }
  float m = red[0];
  __syncthreads();
  float e = __expf(aw[tid] - m);
  aw[tid]  = e;
  red[tid] = e;
  __syncthreads();
  for (int s = 32; s > 0; s >>= 1) {
    if (tid < s) red[tid] += red[tid + s];
    __syncthreads();
  }
  float inv = 1.f / red[0];
  __syncthreads();
  for (int kk = 0; kk < kH / kSenc; ++kk) {
    int k = tid * (kH / kSenc) + kk;
    float c = 0.f;
    for (int s = 0; s < kSenc; ++s) c += aw[s] * (float)eb[(size_t)s * kH + k];
    cat[(size_t)b * (kE + kH) + kE + k] = (bhalf)(c * inv);
    cat[(size_t)b * (kE + kH) + k]      = yemb_t[(size_t)b * ld_ye + k];
  }
}

// ---- argmax over vocab[1:] + 1 -----------------------------------------
__global__ void argmax_rows(const float* __restrict__ scores,
                            float* __restrict__ preds) {
  int row = blockIdx.x;
  const float* s = scores + (size_t)row * kV;
  int tid = threadIdx.x;
  float best = -3.4e38f;
  int bi = kV;
  for (int n = 1 + tid; n < kV; n += blockDim.x) {
    float v = s[n];
    if (v > best) { best = v; bi = n; }
  }
  __shared__ float bv[256];
  __shared__ int   bx[256];
  bv[tid] = best; bx[tid] = bi;
  __syncthreads();
  for (int st = 128; st > 0; st >>= 1) {
    if (tid < st) {
      if (bv[tid + st] > bv[tid] ||
          (bv[tid + st] == bv[tid] && bx[tid + st] < bx[tid])) {
        bv[tid] = bv[tid + st];
        bx[tid] = bx[tid + st];
      }
    }
    __syncthreads();
  }
  if (tid == 0) preds[row] = (float)bx[0];
}

// ------------------------------------------------------------------------
extern "C" void kernel_launch(void* const* d_in, const int* in_sizes, int n_in,
                              void* d_out, int out_size, void* d_ws, size_t ws_size,
                              hipStream_t stream) {
  (void)in_sizes; (void)n_in; (void)out_size; (void)ws_size;
  const int*   xs   = (const int*)d_in[0];
  const int*   ys   = (const int*)d_in[1];
  const float* emb  = (const float*)d_in[2];
  const float *wih_f[4], *whh_f[4], *bih_f[4], *bhh_f[4];
  for (int i = 0; i < 4; ++i) {          // enc0, enc1, dec0, dec1
    wih_f[i] = (const float*)d_in[3 + i * 4 + 0];
    whh_f[i] = (const float*)d_in[3 + i * 4 + 1];
    bih_f[i] = (const float*)d_in[3 + i * 4 + 2];
    bhh_f[i] = (const float*)d_in[3 + i * 4 + 3];
  }
  const float* attnW = (const float*)d_in[19];
  const float* combW = (const float*)d_in[20];

  // ---- workspace carve (all bf16 working set fits in 192MB L2) ----
  char* ws = (char*)d_ws;
  size_t off = 0;
  auto alloc = [&](size_t bytes) -> char* {
    char* p = ws + off;
    off += (bytes + 255) & ~(size_t)255;
    return p;
  };
  bhalf* emb_bf = (bhalf*)alloc((size_t)kV * kE * 2);
  bhalf *wih_bf[4], *whh_bf[4];
  for (int i = 0; i < 4; ++i) {
    wih_bf[i] = (bhalf*)alloc((size_t)kG4H * kH * 2);
    whh_bf[i] = (bhalf*)alloc((size_t)kG4H * kH * 2);
  }
  bhalf* attn_bf = (bhalf*)alloc((size_t)kH * kH * 2);
  bhalf* comb_bf = (bhalf*)alloc((size_t)kE * (kE + kH) * 2);
  bhalf* XA   = (bhalf*)alloc((size_t)kB * kSenc * kE * 2);
  bhalf* XB   = (bhalf*)alloc((size_t)kB * kSenc * kH * 2);
  bhalf* XC   = (bhalf*)alloc((size_t)kB * kSenc * kH * 2);
  bhalf* yemb = (bhalf*)alloc((size_t)kB * kSdec * kE * 2);
  bhalf* hst[2]; float* cst[2];
  for (int l = 0; l < 2; ++l) {
    hst[l] = (bhalf*)alloc((size_t)kB * kH * 2);
    cst[l] = (float*)alloc((size_t)kB * kH * 4);
  }
  bhalf* hid  = (bhalf*)alloc((size_t)kB * kH * 2);
  bhalf* cat  = (bhalf*)alloc((size_t)kB * (kE + kH) * 2);
  bhalf* inp  = (bhalf*)alloc((size_t)kB * kH * 2);
  float* gbuf = (float*)alloc((size_t)kB * kG4H * 4);
  float* neg  = (float*)alloc((size_t)kB * kSenc * 4);

  auto cvt = [&](const float* s, bhalf* d, int n) {
    cvt_bf16<<<(n + 255) / 256, 256, 0, stream>>>(s, d, n);
  };

  // ---- one-shot: bf16 conversion, masks, embedding gathers ----
  cvt(emb, emb_bf, kV * kE);
  for (int i = 0; i < 4; ++i) {
    cvt(wih_f[i], wih_bf[i], kG4H * kH);
    cvt(whh_f[i], whh_bf[i], kG4H * kH);
  }
  cvt(attnW, attn_bf, kH * kH);
  cvt(combW, comb_bf, kE * (kE + kH));
  make_neg<<<(kB * kSenc + 255) / 256, 256, 0, stream>>>(xs, neg);
  embed_gather<<<kB * kSenc, 256, 0, stream>>>(xs, emb_bf, XA);
  embed_gather<<<kB * kSdec, 256, 0, stream>>>(ys, emb_bf, yemb);

  // ---- encoder: 2 layers, scan over time ----
  bhalf* Xin = XA;
  bhalf* Xout = XB;
  for (int l = 0; l < 2; ++l) {
    zero_state<<<(kB * kH + 255) / 256, 256, 0, stream>>>(cst[l], hst[l]);
    for (int t = 0; t < kSenc; ++t) {
      lstm_gates<<<16, 256, 0, stream>>>(Xin + (size_t)t * kE, kSenc * kE,
                                         hst[l], wih_bf[l], kH, whh_bf[l],
                                         bih_f[l], bhh_f[l], gbuf);
      lstm_pointwise<<<(kB * kH + 255) / 256, 256, 0, stream>>>(
          gbuf, cst[l], hst[l], Xout + (size_t)t * kH, kSenc * kH);
    }
    Xin = Xout;
    Xout = XC;
  }
  bhalf* enc_out = Xin;  // top-layer sequence output (XC)

  // ---- decoder: attention + 2 cells + vocab projection per step ----
  float* preds  = (float*)d_out;
  float* scores = (float*)d_out + (size_t)kB * kSdec;
  const int NtH = kH / 16;   // 32 tiles
  const int NtV = kV / 16;   // 2000 tiles
  for (int t = 0; t < kSdec; ++t) {
    // hid = h_top @ attn_W^T
    gemm_bt<1><<<(NtH * 32 + 255) / 256, 256, 0, stream>>>(
        hst[1], kH, attn_bf, kH, NtH, hid, kH);
    // softmax attention + context, cat = [x_emb, ctx]
    attn_ctx<<<kB, kSenc, 0, stream>>>(hid, enc_out, neg,
                                       yemb + (size_t)t * kE, kSdec * kE, cat);
    // inp = tanh(cat @ comb_W^T)
    gemm_bt<2><<<(NtH * 32 + 255) / 256, 256, 0, stream>>>(
        cat, kE + kH, comb_bf, kE + kH, NtH, inp, kH);
    // decoder LSTM layer 0/1 (states continue from encoder finals)
    lstm_gates<<<16, 256, 0, stream>>>(inp, kH, hst[0], wih_bf[2], kH,
                                       whh_bf[2], bih_f[2], bhh_f[2], gbuf);
    lstm_pointwise<<<(kB * kH + 255) / 256, 256, 0, stream>>>(
        gbuf, cst[0], hst[0], (bhalf*)nullptr, 0);
    lstm_gates<<<16, 256, 0, stream>>>(hst[0], kH, hst[1], wih_bf[3], kH,
                                       whh_bf[3], bih_f[3], bhh_f[3], gbuf);
    lstm_pointwise<<<(kB * kH + 255) / 256, 256, 0, stream>>>(
        gbuf, cst[1], hst[1], (bhalf*)nullptr, 0);
    // sc = h_top @ emb^T  (dominant GEMM: 32x32000x512, emb bf16 in L2,
    // each B tile loaded once and used by two WMMAs)
    gemm_bt<0><<<(NtV * 32 + 255) / 256, 256, 0, stream>>>(
        hst[1], kH, emb_bf, kH, NtV, scores + (size_t)t * kV,
        kSdec * kV);
  }

  // preds = argmax(scores[..., 1:]) + 1
  argmax_rows<<<kB * kSdec, 256, 0, stream>>>(scores, preds);
}
